// Bottleneck_DySnake_22385369547469
// MI455X (gfx1250) — matrix operand order repaired and checked by reading, not compiled, not run on hardware
//
#include <hip/hip_runtime.h>
#include <hip/hip_bf16.h>

// Geometry from the reference setup_inputs(): B=8, H=W=80, c1=c2=256, c_=128, K=3
#define BATCH 8
#define IMH   80
#define IMW   80
#define HW    (IMH * IMW)      // 6400
#define NPIX  (BATCH * HW)     // 51200

typedef __attribute__((ext_vector_type(16))) __bf16 v16bf;
typedef __attribute__((ext_vector_type(8)))  float  v8f;

struct U32x8 { uint4 lo, hi; };   // 32-byte blob for v16bf bit_cast

__device__ __forceinline__ unsigned short f32_to_bf16_bits(float f) {
  unsigned int u = __float_as_uint(f);
  u += 0x7FFFu + ((u >> 16) & 1u);   // round-to-nearest-even
  return (unsigned short)(u >> 16);
}
__device__ __forceinline__ unsigned int pack2bf(float a, float b) {
  return (unsigned int)f32_to_bf16_bits(a) | ((unsigned int)f32_to_bf16_bits(b) << 16);
}
__device__ __forceinline__ float silu_f(float x) { return x / (1.0f + expf(-x)); }

// B fragment: LDS tile stored [col][k] with 40-short row stride (80B, 16B aligned).
// Lane's 16 values (K = khalf*16 .. +15 at column `col`) are contiguous 32 bytes.
__device__ __forceinline__ v16bf load_b_frag(const unsigned short* lds, int col, int khalf) {
  const uint4* p = (const uint4*)(lds + col * 40 + khalf * 16);
  U32x8 u; u.lo = p[0]; u.hi = p[1];
  return __builtin_bit_cast(v16bf, u);
}

// ---------------------------------------------------------------------------
// NCHW fp32 -> NHWC fp32 + NHWC bf16, LDS-tiled so both sides are coalesced.
// grid = (HW/32, C/32, B), block = (32,8).
// ---------------------------------------------------------------------------
__global__ void k_nchw_to_nhwc(const float* __restrict__ src,   // [B,C,HW]
                               float* __restrict__ dstf,        // [B,HW,C]
                               unsigned short* __restrict__ dsth,
                               int C) {
  __shared__ float tile[32][33];
  const int b = blockIdx.z;
  const int p0 = blockIdx.x * 32, c0 = blockIdx.y * 32;
  const int tx = threadIdx.x, ty = threadIdx.y;
#pragma unroll
  for (int i = 0; i < 4; ++i)
    tile[ty + 8 * i][tx] = src[(size_t)b * C * HW + (size_t)(c0 + ty + 8 * i) * HW + p0 + tx];
  __syncthreads();
#pragma unroll
  for (int i = 0; i < 4; ++i) {
    float v = tile[tx][ty + 8 * i];
    size_t d = ((size_t)b * HW + p0 + ty + 8 * i) * C + c0 + tx;
    dstf[d] = v;
    dsth[d] = f32_to_bf16_bits(v);
  }
}

// Fused transpose back + residual add: out(NCHW) = x(NCHW) + h(NHWC)
__global__ void k_add_transpose(const float* __restrict__ x,    // [B,C,HW]
                                const float* __restrict__ hf,   // [B,HW,C]
                                float* __restrict__ out, int C) {
  __shared__ float tile[32][33];
  const int b = blockIdx.z;
  const int p0 = blockIdx.x * 32, c0 = blockIdx.y * 32;
  const int tx = threadIdx.x, ty = threadIdx.y;
#pragma unroll
  for (int i = 0; i < 4; ++i)
    tile[tx][ty + 8 * i] = hf[((size_t)b * HW + p0 + ty + 8 * i) * C + c0 + tx];
  __syncthreads();
#pragma unroll
  for (int i = 0; i < 4; ++i) {
    size_t d = (size_t)b * C * HW + (size_t)(c0 + ty + 8 * i) * HW + p0 + tx;
    out[d] = x[d] + tile[ty + 8 * i][tx];
  }
}

// ---------------------------------------------------------------------------
// Weight pack: fp32 weights -> bf16 A fragments in WMMA lane order.
//   mode 0 (conv0): k = q*C + c  -> src = w[m][c][q]   (w: [OC,C,3,3])
//   mode 1 (dsc):   k = kk*C + c -> src = w[m][c][kk]  (w: [OC,C,3])
//   mode 2 (fuse):  k = k        -> src = w[m][k]      (w: [OC,3*OC])
// ---------------------------------------------------------------------------
__global__ void k_pack_afrag(const float* __restrict__ w, unsigned short* __restrict__ dst,
                             int Kt, int nch, int mode, int logC, int total) {
  int d = blockIdx.x * blockDim.x + threadIdx.x;
  if (d >= total) return;
  int f = d >> 9, r = d & 511;
  int lane = r >> 4, j = r & 15;
  int mt = f / nch, chunk = f - mt * nch;
  int khalf = lane >> 4;
  int m = mt * 16 + (lane & 15);
  int k = chunk * 32 + khalf * 16 + j;
  size_t srcidx;
  if (mode == 0) {
    int C = 1 << logC, q = k >> logC, c = k & (C - 1);
    srcidx = ((size_t)m * C + c) * 9 + q;
  } else if (mode == 1) {
    int C = 1 << logC, kk = k >> logC, c = k & (C - 1);
    srcidx = ((size_t)m * C + c) * 3 + kk;
  } else {
    srcidx = (size_t)m * Kt + k;
  }
  dst[d] = f32_to_bf16_bits(w[srcidx]);
}

// ===========================================================================
// GEMM kernels: 256 threads = 8 waves. M-tile 128 (8x16), N-tile 32 (2x16).
// Each wave: one A fragment (2x b128 global), two B fragments (4x ds_load_b128),
// two v_wmma per K-chunk.  grid = (OC/128, NPIX/32).  K is q-major so the
// per-chunk tap/coord/source state is scalar; within a chunk K == channel,
// so staging is contiguous b64 copies and writeback is contiguous per pixel.
// ===========================================================================

// conv0: 3x3 conv as implicit-im2col GEMM (K = C*9); bias + SiLU; writes bf16 NHWC.
__global__ void k_conv3x3_wmma(const unsigned short* __restrict__ xbh,  // [B,HW,C] bf16
                               const unsigned short* __restrict__ apk,
                               const float* __restrict__ bias,
                               unsigned short* __restrict__ outh,       // [B,HW,OC] bf16
                               int C, int logC, int OC) {
  __shared__ __align__(16) unsigned short ldsB[32 * 40];
  const int tid  = threadIdx.x;
  const int lane = tid & 31;
  const int mtg   = blockIdx.x * 8 + (tid >> 5);
  const int Mbase = mtg * 16;
  const int Nbase = blockIdx.y * 32;
  const int nch   = (C * 9) >> 5;

  // staging: thread copies channels [kq*4, kq*4+4) of column n
  const int n  = tid & 31;
  const int kq = tid >> 5;
  const int p  = Nbase + n;
  const int b  = p / HW, hw = p % HW;
  const int h0 = hw / IMW, w0 = hw % IMW;
  const unsigned short* ximg = xbh + (size_t)b * HW * C;

  const int khalf = lane >> 4;
  const uint4* aptr = (const uint4*)(apk + (((size_t)mtg * nch) << 9)) + lane * 2;

  v8f acc0 = {}, acc1 = {};
  for (int ch = 0; ch < nch; ++ch) {
    const int q     = ch >> (logC - 5);            // chunk-uniform filter tap
    const int cbase = (ch << 5) & (C - 1);
    const int dy = q / 3 - 1, dx = q % 3 - 1;
    const int hh = h0 + dy, ww = w0 + dx;
    uint2 val = make_uint2(0u, 0u);
    if ((hh >= 0) & (hh < IMH) & (ww >= 0) & (ww < IMW))
      val = *(const uint2*)(ximg + (size_t)(hh * IMW + ww) * C + cbase + kq * 4);
    *(uint2*)(ldsB + n * 40 + kq * 4) = val;
    __syncthreads();

    const uint4* ap = aptr + (size_t)ch * 64;
    U32x8 au; au.lo = ap[0]; au.hi = ap[1];
    v16bf afrag = __builtin_bit_cast(v16bf, au);
    if (ch + 1 < nch)
      __builtin_prefetch((const void*)(aptr + (size_t)(ch + 1) * 64), 0, 3);
    v16bf b0 = load_b_frag(ldsB, lane & 15, khalf);
    v16bf b1 = load_b_frag(ldsB, 16 + (lane & 15), khalf);
    acc0 = __builtin_amdgcn_wmma_f32_16x16x32_bf16(false, afrag, false, b0, (short)0, acc0, false, false);
    acc1 = __builtin_amdgcn_wmma_f32_16x16x32_bf16(false, afrag, false, b1, (short)0, acc1, false, false);
    __syncthreads();
  }
  const float4* b4 = (const float4*)(bias + Mbase + khalf * 8);
  const float4 bl = b4[0], bh = b4[1];
  const float bv[8] = {bl.x, bl.y, bl.z, bl.w, bh.x, bh.y, bh.z, bh.w};
#pragma unroll
  for (int t = 0; t < 2; ++t) {
    const int pw = Nbase + t * 16 + (lane & 15);
    const v8f& a = t ? acc1 : acc0;
    float v[8];
#pragma unroll
    for (int r = 0; r < 8; ++r) v[r] = silu_f(a[r] + bv[r]);
    uint4 pk;
    pk.x = pack2bf(v[0], v[1]); pk.y = pack2bf(v[2], v[3]);
    pk.z = pack2bf(v[4], v[5]); pk.w = pack2bf(v[6], v[7]);
    *(uint4*)(outh + (size_t)pw * OC + Mbase + khalf * 8) = pk;   // pw already = b*HW+hw
  }
}

// Snake conv: bilinear gather (fp32 NHWC, float4 per tap) fused into staging;
// writes fp32 NHWC pre-GN buffer.
__global__ void k_dsc_wmma(const float* __restrict__ xfh,               // [B,HW,C] fp32
                           const unsigned short* __restrict__ apk,
                           const float* __restrict__ bias,
                           const int* __restrict__ y0a, const int* __restrict__ x0a,
                           const float* __restrict__ wya, const float* __restrict__ wxa,
                           int C, int logC, int OC, int morph,
                           float* __restrict__ out) {                   // [B,HW,OC]
  __shared__ __align__(16) unsigned short ldsB[32 * 40];
  const int tid  = threadIdx.x;
  const int lane = tid & 31;
  const int mtg   = blockIdx.x * 8 + (tid >> 5);
  const int Mbase = mtg * 16;
  const int Nbase = blockIdx.y * 32;
  const int nch   = (C * 3) >> 5;

  const int n  = tid & 31;
  const int kq = tid >> 5;
  const int p  = Nbase + n;
  const int b  = p / HW, hw = p % HW;
  const float* ximg = xfh + (size_t)b * HW * C;
  const int coordbase = ((morph * BATCH + b) * 3) * HW + hw;

  const int khalf = lane >> 4;
  const uint4* aptr = (const uint4*)(apk + (((size_t)mtg * nch) << 9)) + lane * 2;

  v8f acc0 = {}, acc1 = {};
  for (int ch = 0; ch < nch; ++ch) {
    const int kk    = ch >> (logC - 5);            // chunk-uniform snake point
    const int cbase = (ch << 5) & (C - 1);
    const int cidx = coordbase + kk * HW;
    const int y0 = y0a[cidx], x0 = x0a[cidx];
    const float wy = wya[cidx], wx = wxa[cidx];
    const int y1 = min(y0 + 1, IMH - 1), x1 = min(x0 + 1, IMW - 1);
    const float w00 = (1.0f - wy) * (1.0f - wx), w01 = (1.0f - wy) * wx;
    const float w10 = wy * (1.0f - wx),          w11 = wy * wx;
    const int cc = cbase + kq * 4;
    const float4 t00 = *(const float4*)(ximg + (size_t)(y0 * IMW + x0) * C + cc);
    const float4 t01 = *(const float4*)(ximg + (size_t)(y0 * IMW + x1) * C + cc);
    const float4 t10 = *(const float4*)(ximg + (size_t)(y1 * IMW + x0) * C + cc);
    const float4 t11 = *(const float4*)(ximg + (size_t)(y1 * IMW + x1) * C + cc);
    float s0 = w00 * t00.x + w01 * t01.x + w10 * t10.x + w11 * t11.x;
    float s1 = w00 * t00.y + w01 * t01.y + w10 * t10.y + w11 * t11.y;
    float s2 = w00 * t00.z + w01 * t01.z + w10 * t10.z + w11 * t11.z;
    float s3 = w00 * t00.w + w01 * t01.w + w10 * t10.w + w11 * t11.w;
    *(uint2*)(ldsB + n * 40 + kq * 4) = make_uint2(pack2bf(s0, s1), pack2bf(s2, s3));
    __syncthreads();

    const uint4* ap = aptr + (size_t)ch * 64;
    U32x8 au; au.lo = ap[0]; au.hi = ap[1];
    v16bf afrag = __builtin_bit_cast(v16bf, au);
    if (ch + 1 < nch)
      __builtin_prefetch((const void*)(aptr + (size_t)(ch + 1) * 64), 0, 3);
    v16bf b0 = load_b_frag(ldsB, lane & 15, khalf);
    v16bf b1 = load_b_frag(ldsB, 16 + (lane & 15), khalf);
    acc0 = __builtin_amdgcn_wmma_f32_16x16x32_bf16(false, afrag, false, b0, (short)0, acc0, false, false);
    acc1 = __builtin_amdgcn_wmma_f32_16x16x32_bf16(false, afrag, false, b1, (short)0, acc1, false, false);
    __syncthreads();
  }
  const float4* b4 = (const float4*)(bias + Mbase + khalf * 8);
  const float4 bl = b4[0], bh = b4[1];
#pragma unroll
  for (int t = 0; t < 2; ++t) {
    const int pw = Nbase + t * 16 + (lane & 15);
    const v8f& a = t ? acc1 : acc0;
    float* dst = out + (size_t)pw * OC + Mbase + khalf * 8;
    *(float4*)dst       = make_float4(a[0] + bl.x, a[1] + bl.y, a[2] + bl.z, a[3] + bl.w);
    *(float4*)(dst + 4) = make_float4(a[4] + bh.x, a[5] + bh.y, a[6] + bh.z, a[7] + bh.w);
  }
}

// 1x1 fuse over virtual concat(a,bx,by) (all bf16 NHWC): bias + SiLU.
// Writes h in both fp32 NHWC (for next block's gather/offconv + final add)
// and bf16 NHWC (for next block's conv0 staging).
__global__ void k_fuse_wmma(const unsigned short* __restrict__ aP,
                            const unsigned short* __restrict__ bxP,
                            const unsigned short* __restrict__ byP,
                            const unsigned short* __restrict__ apk,
                            const float* __restrict__ bias,
                            int OC, int logOC,
                            float* __restrict__ outf,                // [B,HW,OC]
                            unsigned short* __restrict__ outh) {     // [B,HW,OC]
  __shared__ __align__(16) unsigned short ldsB[32 * 40];
  const int tid  = threadIdx.x;
  const int lane = tid & 31;
  const int mtg   = blockIdx.x * 8 + (tid >> 5);
  const int Mbase = mtg * 16;
  const int Nbase = blockIdx.y * 32;
  const int nch   = (3 * OC) >> 5;

  const int n  = tid & 31;
  const int kq = tid >> 5;
  const int p  = Nbase + n;

  const int khalf = lane >> 4;
  const uint4* aptr = (const uint4*)(apk + (((size_t)mtg * nch) << 9)) + lane * 2;

  v8f acc0 = {}, acc1 = {};
  for (int ch = 0; ch < nch; ++ch) {
    const int part  = ch >> (logOC - 5);           // chunk-uniform source select
    const int cbase = (ch << 5) & (OC - 1);
    const unsigned short* src = (part == 0) ? aP : (part == 1) ? bxP : byP;
    *(uint2*)(ldsB + n * 40 + kq * 4) =
        *(const uint2*)(src + (size_t)p * OC + cbase + kq * 4);
    __syncthreads();

    const uint4* ap = aptr + (size_t)ch * 64;
    U32x8 au; au.lo = ap[0]; au.hi = ap[1];
    v16bf afrag = __builtin_bit_cast(v16bf, au);
    if (ch + 1 < nch)
      __builtin_prefetch((const void*)(aptr + (size_t)(ch + 1) * 64), 0, 3);
    v16bf b0 = load_b_frag(ldsB, lane & 15, khalf);
    v16bf b1 = load_b_frag(ldsB, 16 + (lane & 15), khalf);
    acc0 = __builtin_amdgcn_wmma_f32_16x16x32_bf16(false, afrag, false, b0, (short)0, acc0, false, false);
    acc1 = __builtin_amdgcn_wmma_f32_16x16x32_bf16(false, afrag, false, b1, (short)0, acc1, false, false);
  }
  const float4* b4 = (const float4*)(bias + Mbase + khalf * 8);
  const float4 bl = b4[0], bh = b4[1];
  const float bv[8] = {bl.x, bl.y, bl.z, bl.w, bh.x, bh.y, bh.z, bh.w};
#pragma unroll
  for (int t = 0; t < 2; ++t) {
    const int pw = Nbase + t * 16 + (lane & 15);
    const v8f& a = t ? acc1 : acc0;
    float v[8];
#pragma unroll
    for (int r = 0; r < 8; ++r) v[r] = silu_f(a[r] + bv[r]);
    float* df = outf + (size_t)pw * OC + Mbase + khalf * 8;
    *(float4*)df       = make_float4(v[0], v[1], v[2], v[3]);
    *(float4*)(df + 4) = make_float4(v[4], v[5], v[6], v[7]);
    uint4 pk;
    pk.x = pack2bf(v[0], v[1]); pk.y = pack2bf(v[2], v[3]);
    pk.z = pack2bf(v[4], v[5]); pk.w = pack2bf(v[6], v[7]);
    *(uint4*)(outh + (size_t)pw * OC + Mbase + khalf * 8) = pk;
  }
}

// ---------------------------------------------------------------------------
// Offset conv (2 needed channels) + tanh + snake-cumsum (K=3 => bend =
// [tanh(o0), 0, tanh(o2)]) + clip/floor coordinate precompute.  NHWC input.
// ---------------------------------------------------------------------------
__global__ void k_offcoord(const float* __restrict__ xfh,  // [B,HW,C]
                           const float* __restrict__ ow,   // [6,C,3,3]
                           const float* __restrict__ ob,   // [6]
                           int C, int chanoff, int morph,
                           int* __restrict__ y0a, int* __restrict__ x0a,
                           float* __restrict__ wya, float* __restrict__ wxa) {
  int p = blockIdx.x * blockDim.x + threadIdx.x;
  if (p >= NPIX) return;
  int b = p / HW, hw = p % HW;
  int h = hw / IMW, w = hw % IMW;
  float o0 = ob[chanoff], o2 = ob[chanoff + 2];
  const float* ximg = xfh + (size_t)b * HW * C;
  const float* w0base = ow + (size_t)chanoff * C * 9;
  const float* w2base = ow + (size_t)(chanoff + 2) * C * 9;
#pragma unroll
  for (int q = 0; q < 9; ++q) {
    int hh = h + q / 3 - 1, ww = w + q % 3 - 1;
    if (hh < 0 || hh >= IMH || ww < 0 || ww >= IMW) continue;
    const float* px = ximg + (size_t)(hh * IMW + ww) * C;
    for (int c = 0; c < C; ++c) {
      float xv = px[c];
      o0 += w0base[c * 9 + q] * xv;
      o2 += w2base[c * 9 + q] * xv;
    }
  }
  float bend[3] = {tanhf(o0), 0.0f, tanhf(o2)};
#pragma unroll
  for (int k = 0; k < 3; ++k) {
    float ys, xs;
    if (morph == 0) { ys = (float)h + bend[k]; xs = (float)(w + k - 1); }
    else            { xs = (float)w + bend[k]; ys = (float)(h + k - 1); }
    ys = fminf(fmaxf(ys, 0.0f), (float)(IMH - 1));
    xs = fminf(fmaxf(xs, 0.0f), (float)(IMW - 1));
    float y0f = floorf(ys), x0f = floorf(xs);
    int idx = ((morph * BATCH + b) * 3 + k) * HW + hw;
    y0a[idx] = (int)y0f;  x0a[idx] = (int)x0f;
    wya[idx] = ys - y0f;  wxa[idx] = xs - x0f;
  }
}

// ---------------------------------------------------------------------------
// GroupNorm(groups=OC/4) + ReLU on NHWC: one block per (b, group); each pixel
// contributes one float4 (the group's 4 contiguous channels); bf16 NHWC out.
// ---------------------------------------------------------------------------
__global__ void k_gn_relu(const float* __restrict__ in,    // [B,HW,OC]
                          const float* __restrict__ gma, const float* __restrict__ bta,
                          int OC, unsigned short* __restrict__ out) {  // [B,HW,OC] bf16
  const int ngrp = OC >> 2;
  const int b = blockIdx.x / ngrp, grp = blockIdx.x % ngrp;
  const float* base = in + (size_t)b * HW * OC + grp * 4;
  unsigned short* ob = out + (size_t)b * HW * OC + grp * 4;
  __shared__ float ssum[256], ssq[256];
  float s = 0.0f, q = 0.0f;
  for (int px = threadIdx.x; px < HW; px += 256) {
    const float4 v = *(const float4*)(base + (size_t)px * OC);
    s += v.x + v.y + v.z + v.w;
    q += v.x * v.x + v.y * v.y + v.z * v.z + v.w * v.w;
  }
  ssum[threadIdx.x] = s; ssq[threadIdx.x] = q;
  __syncthreads();
  for (int st = 128; st > 0; st >>= 1) {
    if (threadIdx.x < st) {
      ssum[threadIdx.x] += ssum[threadIdx.x + st];
      ssq[threadIdx.x]  += ssq[threadIdx.x + st];
    }
    __syncthreads();
  }
  const float invn = 1.0f / (4.0f * HW);
  float mean = ssum[0] * invn;
  float var  = ssq[0] * invn - mean * mean;
  float inv  = rsqrtf(var + 1e-5f);
  const float g0 = gma[grp * 4] * inv, g1 = gma[grp * 4 + 1] * inv;
  const float g2 = gma[grp * 4 + 2] * inv, g3 = gma[grp * 4 + 3] * inv;
  const float b0 = bta[grp * 4], b1 = bta[grp * 4 + 1];
  const float b2 = bta[grp * 4 + 2], b3 = bta[grp * 4 + 3];
  for (int px = threadIdx.x; px < HW; px += 256) {
    const float4 v = *(const float4*)(base + (size_t)px * OC);
    float r0 = fmaxf((v.x - mean) * g0 + b0, 0.0f);
    float r1 = fmaxf((v.y - mean) * g1 + b1, 0.0f);
    float r2 = fmaxf((v.z - mean) * g2 + b2, 0.0f);
    float r3 = fmaxf((v.w - mean) * g3 + b3, 0.0f);
    *(uint2*)(ob + (size_t)px * OC) = make_uint2(pack2bf(r0, r1), pack2bf(r2, r3));
  }
}

// ---------------------------------------------------------------------------
// Host orchestration
// ---------------------------------------------------------------------------
extern "C" void kernel_launch(void* const* d_in, const int* in_sizes, int n_in,
                              void* d_out, int out_size, void* d_ws, size_t ws_size,
                              hipStream_t stream) {
  (void)in_sizes; (void)n_in; (void)out_size; (void)ws_size;
  const float* x = (const float*)d_in[0];
  float* out = (float*)d_out;

  char* ws = (char*)d_ws;
  size_t off = 0;
  auto alloc = [&](size_t bytes) -> void* {
    void* p = (void*)(ws + off);
    off += (bytes + 255) & ~(size_t)255;
    return p;
  };
  float*          xfh  = (float*)alloc((size_t)NPIX * 256 * 4);          // x NHWC fp32
  unsigned short* xbh  = (unsigned short*)alloc((size_t)NPIX * 256 * 2); // x NHWC bf16
  unsigned short* wb0  = (unsigned short*)alloc((size_t)256 * 128 * 9 * 2);
  unsigned short* wdsc = (unsigned short*)alloc((size_t)256 * 128 * 3 * 2);
  unsigned short* wfu  = (unsigned short*)alloc((size_t)256 * 768 * 2);
  unsigned short* aH   = (unsigned short*)alloc((size_t)NPIX * 256 * 2);
  unsigned short* bxH  = (unsigned short*)alloc((size_t)NPIX * 256 * 2);
  unsigned short* byH  = (unsigned short*)alloc((size_t)NPIX * 256 * 2);
  float*          pregn= (float*)alloc((size_t)NPIX * 256 * 4);
  float*          h1f  = (float*)alloc((size_t)NPIX * 128 * 4);
  unsigned short* h1h  = (unsigned short*)alloc((size_t)NPIX * 128 * 2);
  float*          h2f  = (float*)alloc((size_t)NPIX * 256 * 4);
  unsigned short* h2h  = (unsigned short*)alloc((size_t)NPIX * 256 * 2);
  int*   y0a = (int*)alloc((size_t)2 * BATCH * 3 * HW * 4);
  int*   x0a = (int*)alloc((size_t)2 * BATCH * 3 * HW * 4);
  float* wya = (float*)alloc((size_t)2 * BATCH * 3 * HW * 4);
  float* wxa = (float*)alloc((size_t)2 * BATCH * 3 * HW * 4);

  auto pack = [&](const float* w, unsigned short* dst, int OC, int Kt, int mode, int logC) {
    int nch = Kt >> 5, total = OC * Kt;
    k_pack_afrag<<<dim3((total + 255) / 256), dim3(256), 0, stream>>>(
        w, dst, Kt, nch, mode, logC, total);
  };

  auto run_block = [&](const float* xf, const unsigned short* xbf16,
                       int C, int logC, int OC, int logOC,
                       void* const* P, float* houtf, unsigned short* houth) {
    const float* conv0_w = (const float*)P[0];
    const float* conv0_b = (const float*)P[1];
    const float* offx_w  = (const float*)P[2];
    const float* offx_b  = (const float*)P[3];
    const float* dscx_w  = (const float*)P[4];
    const float* dscx_b  = (const float*)P[5];
    const float* gnx_g   = (const float*)P[6];
    const float* gnx_b   = (const float*)P[7];
    const float* offy_w  = (const float*)P[8];
    const float* offy_b  = (const float*)P[9];
    const float* dscy_w  = (const float*)P[10];
    const float* dscy_b  = (const float*)P[11];
    const float* gny_g   = (const float*)P[12];
    const float* gny_b   = (const float*)P[13];
    const float* fuse_w  = (const float*)P[14];
    const float* fuse_b  = (const float*)P[15];

    dim3 gg(OC / 128, NPIX / 32), bb(256);
    dim3 gp((NPIX + 255) / 256), bp(256);

    // conv0 + SiLU  -> bf16 NHWC
    pack(conv0_w, wb0, OC, C * 9, 0, logC);
    k_conv3x3_wmma<<<gg, bb, 0, stream>>>(xbf16, wb0, conv0_b, aH, C, logC, OC);

    // snake coordinates for both morphologies
    k_offcoord<<<gp, bp, 0, stream>>>(xf, offx_w, offx_b, C, 0, 0, y0a, x0a, wya, wxa);
    k_offcoord<<<gp, bp, 0, stream>>>(xf, offy_w, offy_b, C, 3, 1, y0a, x0a, wya, wxa);

    // snake conv morph 0 -> GN -> ReLU
    pack(dscx_w, wdsc, OC, C * 3, 1, logC);
    k_dsc_wmma<<<gg, bb, 0, stream>>>(xf, wdsc, dscx_b, y0a, x0a, wya, wxa, C, logC, OC, 0, pregn);
    k_gn_relu<<<dim3(BATCH * (OC / 4)), dim3(256), 0, stream>>>(pregn, gnx_g, gnx_b, OC, bxH);

    // snake conv morph 1 -> GN -> ReLU
    pack(dscy_w, wdsc, OC, C * 3, 1, logC);
    k_dsc_wmma<<<gg, bb, 0, stream>>>(xf, wdsc, dscy_b, y0a, x0a, wya, wxa, C, logC, OC, 1, pregn);
    k_gn_relu<<<dim3(BATCH * (OC / 4)), dim3(256), 0, stream>>>(pregn, gny_g, gny_b, OC, byH);

    // 1x1 fuse + SiLU -> fp32 + bf16 NHWC
    pack(fuse_w, wfu, OC, 3 * OC, 2, logOC);
    k_fuse_wmma<<<gg, bb, 0, stream>>>(aH, bxH, byH, wfu, fuse_b, OC, logOC, houtf, houth);
  };

  // x: NCHW -> NHWC (fp32 + bf16)
  k_nchw_to_nhwc<<<dim3(HW / 32, 256 / 32, BATCH), dim3(32, 8), 0, stream>>>(x, xfh, xbh, 256);

  run_block(xfh, xbh, 256, 8, 128, 7, d_in + 1,  h1f, h1h);
  run_block(h1f, h1h, 128, 7, 256, 8, d_in + 17, h2f, h2h);

  // out = x + h2 (transpose back to NCHW)
  k_add_transpose<<<dim3(HW / 32, 256 / 32, BATCH), dim3(32, 8), 0, stream>>>(x, h2f, out, 256);
}
